// SpatialCrossAttention_2860448219661
// MI455X (gfx1250) — matrix operand order repaired and checked by reading, compiled, not run on hardware
//
#include <hip/hip_runtime.h>
#include <hip/hip_bf16.h>
#include <math.h>

typedef __attribute__((ext_vector_type(16))) _Float16 v16h;
typedef __attribute__((ext_vector_type(8)))  _Float16 v8h;
typedef __attribute__((ext_vector_type(8)))  float    v8f;

#define HEADS 8
#define HD    32
#define NPTS  8
#define ZANCH 4
#define QN    10000
#define NCAM  6
#define DIM   256
#define HF    32
#define WF    88
#define HW    (HF * WF)

// Padded LDS row stride (halves): 40 halves = 80 B = 20 dwords.
// Lane bank base = n*20 mod 64 -> 16 distinct multiples of 4 => conflict-free
// ds_load_b128, and 80/32/16-byte offsets keep every access 16 B aligned.
#define LDT 40

// ---------------------------------------------------------------------------
// WMMA fragment loaders for V_WMMA_F32_16X16X32_F16 (wave32).
// ISA layouts (CDNA5 ISA §7.12.2):
//  A 16x32 f16: lanes 0-15 = rows M=0..15 holding K=0..7,16..23;
//               lanes 16-31 = same rows holding K=8..15,24..31.
//  B 32x16 f16: lane holds column N=lane&15; lanes 0-15 K=0..7,16..23; 16-31 K=8..15,24..31.
//  C/D 16x16 f32: VGPR i -> M = i (+8 for lanes 16-31), N = lane&15.
// Both LDS tiles are stored so a lane's fragment halves are contiguous:
//  As[m][k] (row-major, stride LDT), Bs[n][k] (fragment-major, stride LDT).
// ---------------------------------------------------------------------------
__device__ inline v16h frag_a(const _Float16* As, int mBase) {
  int lane = threadIdx.x & 31;
  const _Float16* row = As + (size_t)(mBase + (lane & 15)) * LDT;
  int kb = (lane < 16) ? 0 : 8;
  v8h lo = *(const v8h*)(row + kb);        // K = kb..kb+7
  v8h hi = *(const v8h*)(row + kb + 16);   // K = kb+16..kb+23
  return __builtin_shufflevector(lo, hi, 0, 1, 2, 3, 4, 5, 6, 7,
                                 8, 9, 10, 11, 12, 13, 14, 15);
}

__device__ inline v16h frag_b(const _Float16* Bs, int nBase) {
  int lane = threadIdx.x & 31;
  const _Float16* col = Bs + (size_t)(nBase + (lane & 15)) * LDT;
  int kb = (lane < 16) ? 0 : 8;
  v8h lo = *(const v8h*)(col + kb);        // K = kb..kb+7
  v8h hi = *(const v8h*)(col + kb + 16);   // K = kb+16..kb+23
  return __builtin_shufflevector(lo, hi, 0, 1, 2, 3, 4, 5, 6, 7,
                                 8, 9, 10, 11, 12, 13, 14, 15);
}

// One K=32 step: 4 waves, wave w computes rows [16w,16w+16) x 64 cols.
// All fragments are loaded into independent registers first so the DS loads
// can overlap WMMA issue instead of serializing on a reused register range.
__device__ inline void mma_step(const _Float16* As, const _Float16* Bs, v8f acc[4]) {
  int wave = threadIdx.x >> 5;
  v16h a  = frag_a(As, wave * 16);
  v16h b0 = frag_b(Bs, 0);
  v16h b1 = frag_b(Bs, 16);
  v16h b2 = frag_b(Bs, 32);
  v16h b3 = frag_b(Bs, 48);
  acc[0] = __builtin_amdgcn_wmma_f32_16x16x32_f16(false, a, false, b0, (short)0, acc[0], false, false);
  acc[1] = __builtin_amdgcn_wmma_f32_16x16x32_f16(false, a, false, b1, (short)0, acc[1], false, false);
  acc[2] = __builtin_amdgcn_wmma_f32_16x16x32_f16(false, a, false, b2, (short)0, acc[2], false, false);
  acc[3] = __builtin_amdgcn_wmma_f32_16x16x32_f16(false, a, false, b3, (short)0, acc[3], false, false);
}

// Bs[n][k] = W[(n0+n)*ldW + k0+k]  (W row-major [N][K]); 128 threads.
// Contiguous 16-half writes -> ds_store_b128 x2.
__device__ inline void load_b_weights(_Float16* Bs, const float* W, int n0, int k0, int ldW) {
  int t = threadIdx.x;
  int n = t >> 1;            // 0..63
  int kh = (t & 1) * 16;     // 0 or 16
  const float* src = W + (size_t)(n0 + n) * ldW + k0 + kh;
  v8h t0, t1;
#pragma unroll
  for (int j = 0; j < 8; ++j) t0[j] = (_Float16)src[j];
#pragma unroll
  for (int j = 0; j < 8; ++j) t1[j] = (_Float16)src[8 + j];
  _Float16* dst = Bs + (size_t)n * LDT + kh;
  *(v8h*)dst = t0;
  *(v8h*)(dst + 8) = t1;
}

// As[m][k] = A[row][k0+k] (+ A2[row][k0+k]); zero-pad rows >= M; 128 threads.
__device__ inline void load_a_rowmajor(_Float16* As, const float* A, const float* A2,
                                       int m0, int k0, int M, int ldA) {
  int t = threadIdx.x;
  int m = t >> 1;
  int kh = (t & 1) * 16;
  int row = m0 + m;
  _Float16* dst = As + (size_t)m * LDT + kh;
  v8h t0, t1;
  if (row < M) {
    const float* src = A + (size_t)row * ldA + k0 + kh;
    // speculative prefetch of this row's next K-chunk (dropped if OOB)
    __builtin_prefetch(src + 32, 0, 0);
    if (A2) {
      const float* src2 = A2 + (size_t)row * ldA + k0 + kh;
      __builtin_prefetch(src2 + 32, 0, 0);
#pragma unroll
      for (int j = 0; j < 8; ++j) t0[j] = (_Float16)(src[j] + src2[j]);
#pragma unroll
      for (int j = 0; j < 8; ++j) t1[j] = (_Float16)(src[8 + j] + src2[8 + j]);
    } else {
#pragma unroll
      for (int j = 0; j < 8; ++j) t0[j] = (_Float16)src[j];
#pragma unroll
      for (int j = 0; j < 8; ++j) t1[j] = (_Float16)src[8 + j];
    }
  } else {
#pragma unroll
    for (int j = 0; j < 8; ++j) { t0[j] = (_Float16)0.f; t1[j] = (_Float16)0.f; }
  }
  *(v8h*)dst = t0;
  *(v8h*)(dst + 8) = t1;
}

// ---------------------------------------------------------------------------
// Kernel 1: value = (feat(bn,d,hw)^T + lvl + cam) @ Wv^T + bv
// Output layout value_t[bn][head][hw][hd] for the sampler.
// ---------------------------------------------------------------------------
__global__ void __launch_bounds__(128)
value_gemm(const float* __restrict__ feat, const float* __restrict__ lvl,
           const float* __restrict__ cam, const float* __restrict__ Wv,
           const float* __restrict__ bv, float* __restrict__ value_t) {
  __shared__ alignas(16) _Float16 As[64 * LDT];
  __shared__ alignas(16) _Float16 Bs[64 * LDT];
  int bn = blockIdx.z;
  int hw0 = blockIdx.x * 64;
  int n0 = blockIdx.y * 64;
  v8f acc[4] = {};
  for (int k0 = 0; k0 < DIM; k0 += 32) {
    // A: transpose-read of features + embeddings.
    // Thread t owns (m = t>>1, k-half = (t&1)*16): for each j the 64 threads
    // of a half-group read 64 consecutive hw addresses (coalesced), and the
    // LDS write is 16 contiguous halves -> ds_store_b128 x2.
    {
      int t = threadIdx.x;
      int m = t >> 1;            // 0..63  (hw row within tile)
      int kh = (t & 1) * 16;     // 0 or 16
      const float* src = feat + ((size_t)bn * DIM + (k0 + kh)) * HW + hw0 + m;
      const float* lv = lvl + k0 + kh;
      const float* cm = cam + bn * DIM + k0 + kh;
      v8h t0, t1;
#pragma unroll
      for (int j = 0; j < 8; ++j)
        t0[j] = (_Float16)(src[(size_t)j * HW] + lv[j] + cm[j]);
#pragma unroll
      for (int j = 0; j < 8; ++j)
        t1[j] = (_Float16)(src[(size_t)(8 + j) * HW] + lv[8 + j] + cm[8 + j]);
      _Float16* dst = As + (size_t)m * LDT + kh;
      *(v8h*)dst = t0;
      *(v8h*)(dst + 8) = t1;
    }
    load_b_weights(Bs, Wv, n0, k0, DIM);
    __syncthreads();
    mma_step(As, Bs, acc);
    __syncthreads();
  }
  int lane = threadIdx.x & 31;
  int wave = threadIdx.x >> 5;
  int mrow = hw0 + wave * 16 + ((lane < 16) ? 0 : 8);
#pragma unroll
  for (int t = 0; t < 4; ++t) {
    int n = n0 + t * 16 + (lane & 15);
    int head = n >> 5, hd = n & 31;
    float bias = bv[n];
    float* dst = value_t + (((size_t)bn * HEADS + head) * HW) * HD + hd;
#pragma unroll
    for (int i = 0; i < 8; ++i) {
      int m = mrow + i;
      dst[(size_t)m * HD] = acc[t][i] + bias;
    }
  }
}

// ---------------------------------------------------------------------------
// Kernel 2: projRaw[r][0:128]=qp@W_off^T+b_off, [128:192]=qp@W_attn^T+b_attn
// qp = queries + pos_emb fused into A-tile load. M = 60000, N = 192.
// ---------------------------------------------------------------------------
__global__ void __launch_bounds__(128)
proj_gemm(const float* __restrict__ queries, const float* __restrict__ pos,
          const float* __restrict__ Woff, const float* __restrict__ boff,
          const float* __restrict__ Wattn, const float* __restrict__ battn,
          float* __restrict__ projRaw) {
  __shared__ alignas(16) _Float16 As[64 * LDT];
  __shared__ alignas(16) _Float16 Bs[64 * LDT];
  const int M = NCAM * QN;
  int m0 = blockIdx.x * 64;
  int n0 = blockIdx.y * 64;  // 0, 64 -> W_off ; 128 -> W_attn
  const float* W = (n0 < 128) ? Woff : Wattn;
  int nW0 = (n0 < 128) ? n0 : (n0 - 128);
  v8f acc[4] = {};
  for (int k0 = 0; k0 < DIM; k0 += 32) {
    load_a_rowmajor(As, queries, pos, m0, k0, M, DIM);
    load_b_weights(Bs, W, nW0, k0, DIM);
    __syncthreads();
    mma_step(As, Bs, acc);
    __syncthreads();
  }
  int lane = threadIdx.x & 31;
  int wave = threadIdx.x >> 5;
  int mbase = m0 + wave * 16 + ((lane < 16) ? 0 : 8);
#pragma unroll
  for (int t = 0; t < 4; ++t) {
    int n = n0 + t * 16 + (lane & 15);
    float bias = (n < 128) ? boff[n] : battn[n - 128];
#pragma unroll
    for (int i = 0; i < 8; ++i) {
      int m = mbase + i;
      if (m < M) projRaw[(size_t)m * 192 + n] = acc[t][i] + bias;
    }
  }
}

// ---------------------------------------------------------------------------
// Kernel 3: deformable sampling. Block = (bn,q); wave = head; lane = channel.
// Softmax over 8 points per head, 4-corner bilinear gather from value_t
// (17 MB -> L2-resident on a 192 MB L2).
// ---------------------------------------------------------------------------
__global__ void __launch_bounds__(256)
sample_kernel(const float* __restrict__ projRaw, const float* __restrict__ ref3d,
              const float* __restrict__ value_t, float* __restrict__ msdaOut) {
  int b = blockIdx.x;              // 0..NCAM*QN-1
  int bn = b / QN;
  int head = threadIdx.x >> 5;
  int lane = threadIdx.x & 31;
  const float* pr = projRaw + (size_t)b * 192;

  float lg[NPTS];
  float mx = -1e30f;
#pragma unroll
  for (int p = 0; p < NPTS; ++p) {
    lg[p] = pr[128 + head * NPTS + p];
    mx = fmaxf(mx, lg[p]);
  }
  float s = 0.f;
#pragma unroll
  for (int p = 0; p < NPTS; ++p) { lg[p] = __expf(lg[p] - mx); s += lg[p]; }
  float inv = 1.f / s;

  const float* vhead = value_t + ((size_t)bn * HEADS + head) * (size_t)HW * HD + lane;
  const float* rf = ref3d + (size_t)b * ZANCH * 2;
  float acc = 0.f;
#pragma unroll
  for (int p = 0; p < NPTS; ++p) {
    int z = p & (ZANCH - 1);  // p = pz*ZANCH + z
    float ox = pr[(head * NPTS + p) * 2 + 0];
    float oy = pr[(head * NPTS + p) * 2 + 1];
    float lx = rf[z * 2 + 0] + ox * (1.0f / WF);
    float ly = rf[z * 2 + 1] + oy * (1.0f / HF);
    float x = lx * WF - 0.5f;
    float y = ly * HF - 0.5f;
    float xf = floorf(x), yf = floorf(y);
    float dx = x - xf, dy = y - yf;
    int x0 = (int)xf, y0 = (int)yf;
    float wp = lg[p] * inv;
#pragma unroll
    for (int cy = 0; cy < 2; ++cy) {
#pragma unroll
      for (int cx = 0; cx < 2; ++cx) {
        int xi = x0 + cx, yi = y0 + cy;
        if (xi >= 0 && xi < WF && yi >= 0 && yi < HF) {
          float cw = (cx ? dx : 1.f - dx) * (cy ? dy : 1.f - dy);
          acc += wp * cw * vhead[(size_t)(yi * WF + xi) * HD];
        }
      }
    }
  }
  msdaOut[(size_t)b * DIM + head * HD + lane] = acc;
}

// ---------------------------------------------------------------------------
// Kernel 4/6: plain GEMM out = A @ W^T + bias  (K = N = 256)
// ---------------------------------------------------------------------------
__global__ void __launch_bounds__(128)
dense_gemm(const float* __restrict__ A, const float* __restrict__ W,
           const float* __restrict__ bias, float* __restrict__ out, int M) {
  __shared__ alignas(16) _Float16 As[64 * LDT];
  __shared__ alignas(16) _Float16 Bs[64 * LDT];
  int m0 = blockIdx.x * 64;
  int n0 = blockIdx.y * 64;
  v8f acc[4] = {};
  for (int k0 = 0; k0 < DIM; k0 += 32) {
    load_a_rowmajor(As, A, nullptr, m0, k0, M, DIM);
    load_b_weights(Bs, W, n0, k0, DIM);
    __syncthreads();
    mma_step(As, Bs, acc);
    __syncthreads();
  }
  int lane = threadIdx.x & 31;
  int wave = threadIdx.x >> 5;
  int mbase = m0 + wave * 16 + ((lane < 16) ? 0 : 8);
#pragma unroll
  for (int t = 0; t < 4; ++t) {
    int n = n0 + t * 16 + (lane & 15);
    float bv = bias[n];
#pragma unroll
    for (int i = 0; i < 8; ++i) {
      int m = mbase + i;
      if (m < M) out[(size_t)m * DIM + n] = acc[t][i] + bv;
    }
  }
}

// ---------------------------------------------------------------------------
// Kernel 5: camera scatter/normalize: slots[q][d] = sum_n attn*hit / max(cnt,1)
// ---------------------------------------------------------------------------
__global__ void __launch_bounds__(256)
reduce_cam(const float* __restrict__ innerOut, const unsigned char* __restrict__ bev_mask,
           float* __restrict__ slots) {
  int idx = blockIdx.x * blockDim.x + threadIdx.x;
  if (idx >= QN * DIM) return;
  int q = idx >> 8;
  int d = idx & 255;
  float sum = 0.f, cnt = 0.f;
#pragma unroll
  for (int n = 0; n < NCAM; ++n) {
    const unsigned char* mrow = bev_mask + ((size_t)(n * QN + q) * ZANCH) * 2;
    int h = mrow[0] | mrow[2] | mrow[4] | mrow[6];  // any z of [...,0]
    if (h) {
      sum += innerOut[((size_t)n * QN + q) * DIM + d];
      cnt += 1.f;
    }
  }
  slots[idx] = sum / fmaxf(cnt, 1.f);
}

// ---------------------------------------------------------------------------
extern "C" void kernel_launch(void* const* d_in, const int* in_sizes, int n_in,
                              void* d_out, int out_size, void* d_ws, size_t ws_size,
                              hipStream_t stream) {
  (void)in_sizes; (void)n_in; (void)out_size; (void)ws_size;
  const float* queries   = (const float*)d_in[0];
  const float* pos_emb   = (const float*)d_in[1];
  const float* lvl_emb   = (const float*)d_in[2];
  const float* cam_emb   = (const float*)d_in[3];
  const float* features0 = (const float*)d_in[4];
  const float* ref3d     = (const float*)d_in[5];
  const unsigned char* bev_mask = (const unsigned char*)d_in[6];
  const float* W_value = (const float*)d_in[7];
  const float* b_value = (const float*)d_in[8];
  const float* W_off   = (const float*)d_in[9];
  const float* b_off   = (const float*)d_in[10];
  const float* W_attn  = (const float*)d_in[11];
  const float* b_attn  = (const float*)d_in[12];
  const float* W_inner = (const float*)d_in[13];
  const float* b_inner = (const float*)d_in[14];
  const float* W_out   = (const float*)d_in[15];
  const float* b_out   = (const float*)d_in[16];

  // Workspace layout (floats)
  float* ws       = (float*)d_ws;
  float* value_t  = ws;                                        // 4,325,376
  float* projRaw  = value_t + (size_t)NCAM * HEADS * HW * HD;  // 11,520,000
  float* msdaOut  = projRaw + (size_t)NCAM * QN * 192;         // 15,360,000
  float* innerOut = msdaOut + (size_t)NCAM * QN * DIM;         // 15,360,000
  float* slots    = projRaw;  // projRaw dead after sampling; reuse

  const int M = NCAM * QN;
  dim3 blk(128);

  value_gemm<<<dim3(HW / 64, DIM / 64, NCAM), blk, 0, stream>>>(
      features0, lvl_emb, cam_emb, W_value, b_value, value_t);

  proj_gemm<<<dim3((M + 63) / 64, 3), blk, 0, stream>>>(
      queries, pos_emb, W_off, b_off, W_attn, b_attn, projRaw);

  sample_kernel<<<M, 256, 0, stream>>>(projRaw, ref3d, value_t, msdaOut);

  dense_gemm<<<dim3((M + 63) / 64, DIM / 64), blk, 0, stream>>>(
      msdaOut, W_inner, b_inner, innerOut, M);

  reduce_cam<<<(QN * DIM + 255) / 256, 256, 0, stream>>>(innerOut, bev_mask, slots);

  dense_gemm<<<dim3((QN + 63) / 64, DIM / 64), blk, 0, stream>>>(
      slots, W_out, b_out, (float*)d_out, QN);
}
